// DLRM_Net_42202348650761
// MI455X (gfx1250) — matrix operand order repaired and checked by reading, not compiled, not run on hardware
//
#include <hip/hip_runtime.h>

// ---------------------------------------------------------------------------
// DLRM forward for MI455X (gfx1250, wave32, WMMA bf16 -> f32 accumulate)
// ---------------------------------------------------------------------------

typedef __attribute__((ext_vector_type(16))) __bf16 v16bf;
typedef __attribute__((ext_vector_type(8)))  float  v8f;

#define BATCH   16384
#define N_TAB   26
#define N_ROWS  200000
#define M_SPA   30
#define R_DIM   381   // 30 + 27*26/2

// round-to-nearest-even f32 -> bf16 bit pattern
__device__ inline unsigned short bf16bits(float f) {
    unsigned int u = __builtin_bit_cast(unsigned int, f);
    u += 0x7FFFu + ((u >> 16) & 1u);
    return (unsigned short)(u >> 16);
}
__device__ inline __bf16 f2bf(float f) {
    unsigned short h = bf16bits(f);
    return __builtin_bit_cast(__bf16, h);
}
__device__ inline unsigned int pack2bf(float a, float b) {
    return (unsigned int)bf16bits(a) | ((unsigned int)bf16bits(b) << 16);
}

// A fragment: 16(M) x 32(K), source row-major [16+][ldk] bf16 in LDS.
// ISA 7.12.2 16-bit A layout: lane<16 -> M=lane, VGPR v<4: K=2v,2v+1, v>=4: K=16+2(v-4)+{0,1}
//                             lane>=16 -> same rows, K offset +8
__device__ inline v16bf load_frag_a(const __bf16* p, int ldk) {
    int lane = threadIdx.x & 31;
    int row  = lane & 15;
    int ko   = (lane < 16) ? 0 : 8;
    v16bf a;
#pragma unroll
    for (int v = 0; v < 8; ++v) {
        int k = ((v < 4) ? (2 * v) : (16 + 2 * (v - 4))) + ko;
        a[2 * v]     = p[row * ldk + k];
        a[2 * v + 1] = p[row * ldk + k + 1];
    }
    return a;
}

// B fragment: 32(K) x 16(N) with LDS holding B^T (i.e. [n][k], row-major, stride ldk).
// ISA layout: N=lane%16; VGPR v<4: K=2v,2v+1, v>=4: K=16+2(v-4)+{0,1}; lane>=16 -> K offset +8
__device__ inline v16bf load_frag_b(const __bf16* p, int ldk) {
    int lane = threadIdx.x & 31;
    int n    = lane & 15;
    int ko   = (lane < 16) ? 0 : 8;
    v16bf b;
#pragma unroll
    for (int v = 0; v < 8; ++v) {
        int k = ((v < 4) ? (2 * v) : (16 + 2 * (v - 4))) + ko;
        b[2 * v]     = p[n * ldk + k];
        b[2 * v + 1] = p[n * ldk + k + 1];
    }
    return b;
}

// ---------------------------------------------------------------------------
// Fused GEMM: Y[M][N] = relu(X[M][K] * W[N][K]^T + bias[N])
// Block tile 128x128, 8 waves (2 along M x 4 along N), wave tile 64x32
// = 4x2 WMMA tiles = 8 v_wmma per K-step.
// Software-pipelined: global loads for tile k+1 are issued into registers
// before the WMMAs on tile k (LDS ping-pong, ONE barrier per K-step).
// ---------------------------------------------------------------------------
#define BM 128
#define BN 128
#define BK 32

__global__ __launch_bounds__(256) void gemm_bias_relu(
    const float* __restrict__ X, const float* __restrict__ W,
    const float* __restrict__ bias, float* __restrict__ Y,
    int M, int N, int K)
{
    __shared__ __bf16 As[2][BM][BK];
    __shared__ __bf16 Bs[2][BN][BK];

    const int tid  = threadIdx.x;
    const int wave = tid >> 5;
    const int lane = tid & 31;
    const int wm   = (wave & 1) * 64;   // 2 waves along M, 64 rows each
    const int wn   = (wave >> 1) * 32;  // 4 waves along N, 32 cols each
    const long m0  = (long)blockIdx.y * BM;
    const long n0  = (long)blockIdx.x * BN;
    const bool fullN = (n0 + BN <= (long)N);

    const v8f vzero = {0.f, 0.f, 0.f, 0.f, 0.f, 0.f, 0.f, 0.f};
    v8f acc[4][2];
#pragma unroll
    for (int mi = 0; mi < 4; ++mi)
#pragma unroll
        for (int ni = 0; ni < 2; ++ni) acc[mi][ni] = vzero;

    float ra[16], rb[16];   // register-held staging for the next tile

    // quad mapping shared by both load paths: j = q*256+tid,
    // row = j>>3 (0..127), kq = (j&7)*4 (0..28)
    auto load_tile = [&](int k0) {
        const bool fast = fullN && (k0 + BK <= K);
        if (fast) {
#pragma unroll
            for (int q = 0; q < 4; ++q) {
                int j = q * 256 + tid, row = j >> 3, kq = (j & 7) * 4;
                const float4 va =
                    *(const float4*)(X + (m0 + row) * (long)K + k0 + kq);
                ra[4*q+0] = va.x; ra[4*q+1] = va.y; ra[4*q+2] = va.z; ra[4*q+3] = va.w;
                const float4 vb =
                    *(const float4*)(W + (n0 + row) * (long)K + k0 + kq);
                rb[4*q+0] = vb.x; rb[4*q+1] = vb.y; rb[4*q+2] = vb.z; rb[4*q+3] = vb.w;
                if (q == 0 && k0 + 2 * BK <= K)  // prefetch tile after next
                    __builtin_prefetch(X + (m0 + row) * (long)K + k0 + BK + kq, 0, 3);
            }
        } else {
            // branchless: clamped address + select-to-zero (no exec churn)
#pragma unroll
            for (int q = 0; q < 4; ++q) {
                int j = q * 256 + tid, row = j >> 3, kq = (j & 7) * 4;
#pragma unroll
                for (int t = 0; t < 4; ++t) {
                    int  kk = k0 + kq + t;
                    int  kc = (kk < K) ? kk : (K - 1);
                    float va = X[(m0 + row) * (long)K + kc];
                    ra[4*q+t] = (kk < K) ? va : 0.f;
                    long nn = n0 + row;
                    long nc = (nn < (long)N) ? nn : (long)(N - 1);
                    float vb = W[nc * (long)K + kc];
                    rb[4*q+t] = (kk < K && nn < (long)N) ? vb : 0.f;
                }
            }
        }
    };
    auto store_tile = [&](int pp) {
#pragma unroll
        for (int q = 0; q < 4; ++q) {
            int j = q * 256 + tid, row = j >> 3, kq = (j & 7) * 4;
            unsigned int* pa = (unsigned int*)&As[pp][row][kq];
            pa[0] = pack2bf(ra[4*q+0], ra[4*q+1]);
            pa[1] = pack2bf(ra[4*q+2], ra[4*q+3]);
            unsigned int* pb = (unsigned int*)&Bs[pp][row][kq];
            pb[0] = pack2bf(rb[4*q+0], rb[4*q+1]);
            pb[1] = pack2bf(rb[4*q+2], rb[4*q+3]);
        }
    };

    // prologue: stage tile 0 into buffer 0
    load_tile(0);
    store_tile(0);
    int p = 0;

    for (int k0 = 0; k0 < K; k0 += BK) {
        __syncthreads();                    // buffer p ready; prior reads done
        const bool has_next = (k0 + BK) < K;
        if (has_next) load_tile(k0 + BK);   // global loads overlap the WMMAs

        v16bf a[4], b[2];
#pragma unroll
        for (int mi = 0; mi < 4; ++mi)
            a[mi] = load_frag_a(&As[p][wm + mi * 16][0], BK);
#pragma unroll
        for (int ni = 0; ni < 2; ++ni)
            b[ni] = load_frag_b(&Bs[p][wn + ni * 16][0], BK);
#pragma unroll
        for (int mi = 0; mi < 4; ++mi)
#pragma unroll
            for (int ni = 0; ni < 2; ++ni)
                acc[mi][ni] = __builtin_amdgcn_wmma_f32_16x16x32_bf16(
                    false, a[mi], false, b[ni], (short)0, acc[mi][ni], false, false);

        if (has_next) store_tile(p ^ 1);    // fill the other buffer
        p ^= 1;
    }

    // epilogue: D layout -> lane<16: N=lane, M=r ; lane>=16: N=lane-16, M=8+r
    const int nl = lane & 15;
    const int mb = (lane < 16) ? 0 : 8;
    float bia[2];
#pragma unroll
    for (int ni = 0; ni < 2; ++ni) {
        long n = n0 + wn + ni * 16 + nl;
        bia[ni] = bias[(n < (long)N) ? n : (long)(N - 1)];
    }
#pragma unroll
    for (int mi = 0; mi < 4; ++mi)
#pragma unroll
        for (int ni = 0; ni < 2; ++ni)
#pragma unroll
            for (int r = 0; r < 8; ++r) {
                long n = n0 + wn + ni * 16 + nl;
                if (n < (long)N) {
                    long m = m0 + wm + mi * 16 + mb + r;
                    float v = fmaxf(acc[mi][ni][r] + bia[ni], 0.f);
                    Y[m * (long)N + n] = v;
                }
            }
}

// ---------------------------------------------------------------------------
// Fused embedding gather + pairwise interaction.
// One wave per sample: stage T (27x30 padded to 32x32 bf16) in LDS,
// Z = T * T^T via 4 WMMAs, scatter tril(k=-1) into R[:,30:381]; R[:,0:30]=x.
// ---------------------------------------------------------------------------
__global__ __launch_bounds__(256) void gather_interact(
    const float* __restrict__ bot,   // [B][30]
    const float* __restrict__ emb,   // [26][200000][30]
    const int*   __restrict__ lS_i,  // [26][B]
    float* __restrict__ R)           // [B][381]
{
    __shared__ __bf16 T[8][32][32];

    const int tid  = threadIdx.x;
    const int wave = tid >> 5;
    const int lane = tid & 31;
    const long s   = (long)blockIdx.x * 8 + wave;

    // each lane stages one row of T; rows >26 read a valid dummy row and are
    // zeroed by select (branchless, no exec-mask churn)
    const bool valid = (lane <= N_TAB);
    const int  t     = (lane >= 1 && lane <= N_TAB) ? (lane - 1) : 0;
    const long idx   = (long)lS_i[(long)t * BATCH + s];
    const float* src = (lane == 0) ? (bot + s * M_SPA)
                                   : (emb + ((long)t * N_ROWS + idx) * M_SPA);
    // rows are 30 f32 = 120 B apart -> 8-byte aligned: use float2 loads
#pragma unroll
    for (int c = 0; c < M_SPA; c += 2) {
        const float2 v2 = *(const float2*)(src + c);
        float v0 = valid ? v2.x : 0.f;
        float v1 = valid ? v2.y : 0.f;
        *(unsigned int*)&T[wave][lane][c] = pack2bf(v0, v1);
    }
    *(unsigned int*)&T[wave][lane][M_SPA] = 0u;   // pad cols 30,31

    // R[:,0:30] = bottom-MLP output (f32, full precision)
    if (lane < M_SPA) R[s * R_DIM + lane] = bot[s * M_SPA + lane];
    __syncthreads();

    v16bf a0 = load_frag_a(&T[wave][0][0],  32);
    v16bf a1 = load_frag_a(&T[wave][16][0], 32);
    v16bf b0 = load_frag_b(&T[wave][0][0],  32);  // B[k][j] = T[j][k]
    v16bf b1 = load_frag_b(&T[wave][16][0], 32);

    const v8f vzero = {0.f, 0.f, 0.f, 0.f, 0.f, 0.f, 0.f, 0.f};
    v8f z[2][2];
    z[0][0] = __builtin_amdgcn_wmma_f32_16x16x32_bf16(false, a0, false, b0, (short)0, vzero, false, false);
    z[0][1] = __builtin_amdgcn_wmma_f32_16x16x32_bf16(false, a0, false, b1, (short)0, vzero, false, false);
    z[1][0] = __builtin_amdgcn_wmma_f32_16x16x32_bf16(false, a1, false, b0, (short)0, vzero, false, false);
    z[1][1] = __builtin_amdgcn_wmma_f32_16x16x32_bf16(false, a1, false, b1, (short)0, vzero, false, false);

    const int nl = lane & 15;
    const int mb = (lane < 16) ? 0 : 8;
#pragma unroll
    for (int mi = 0; mi < 2; ++mi)
#pragma unroll
        for (int ni = 0; ni < 2; ++ni)
#pragma unroll
            for (int r = 0; r < 8; ++r) {
                int i = mi * 16 + mb + r;   // row
                int j = ni * 16 + nl;       // col
                if (i < 27 && j < i)
                    R[s * R_DIM + M_SPA + (i * (i - 1)) / 2 + j] = z[mi][ni][r];
            }
}

// ---------------------------------------------------------------------------
// Final layer: out[m] = sigmoid(dot(X[m][0:256], w) + b)
// ---------------------------------------------------------------------------
__global__ __launch_bounds__(256) void matvec_sigmoid(
    const float* __restrict__ X, const float* __restrict__ w,
    const float* __restrict__ b, float* __restrict__ out, int M, int K)
{
    int m = blockIdx.x * blockDim.x + threadIdx.x;
    if (m >= M) return;
    float s = b[0];
    for (int k = 0; k < K; ++k) s += X[(long)m * K + k] * w[k];
    out[m] = 1.f / (1.f + expf(-s));
}

// ---------------------------------------------------------------------------
// Launch: bottom MLP -> gather+interact -> top MLP -> matvec+sigmoid
// d_in order: dense_x, lS_o, lS_i, emb_tables, bot_ws[0..3], bot_bs[0..3],
//             top_ws[0..4], top_bs[0..4]
// ---------------------------------------------------------------------------
extern "C" void kernel_launch(void* const* d_in, const int* in_sizes, int n_in,
                              void* d_out, int out_size, void* d_ws, size_t ws_size,
                              hipStream_t stream) {
    (void)in_sizes; (void)n_in; (void)out_size; (void)ws_size;

    const float* dense_x = (const float*)d_in[0];
    const int*   lS_i    = (const int*)d_in[2];
    const float* emb     = (const float*)d_in[3];
    const float* bw[4]   = {(const float*)d_in[4],  (const float*)d_in[5],
                            (const float*)d_in[6],  (const float*)d_in[7]};
    const float* bb[4]   = {(const float*)d_in[8],  (const float*)d_in[9],
                            (const float*)d_in[10], (const float*)d_in[11]};
    const float* tw[5]   = {(const float*)d_in[12], (const float*)d_in[13],
                            (const float*)d_in[14], (const float*)d_in[15],
                            (const float*)d_in[16]};
    const float* tb[5]   = {(const float*)d_in[17], (const float*)d_in[18],
                            (const float*)d_in[19], (const float*)d_in[20],
                            (const float*)d_in[21]};
    float* out = (float*)d_out;

    // workspace carve-up (~123 MB)
    char*  ws      = (char*)d_ws;
    float* buf0    = (float*)ws;                                   // 16384*1024 f32
    float* buf1    = (float*)(ws + (size_t)BATCH * 1024 * 4);      // 16384*512  f32
    float* bot_out = (float*)(ws + (size_t)BATCH * 1536 * 4);      // 16384*30   f32
    float* Rbuf    = (float*)(ws + (size_t)BATCH * 1600 * 4);      // 16384*381  f32

    dim3 blk(256);
    auto gemm = [&](const float* X, const float* W, const float* bias,
                    float* Y, int M, int N, int K) {
        dim3 grid((N + BN - 1) / BN, (M + BM - 1) / BM);
        gemm_bias_relu<<<grid, blk, 0, stream>>>(X, W, bias, Y, M, N, K);
    };

    // bottom MLP: 4 -> 512 -> 256 -> 128 -> 30 (all ReLU)
    gemm(dense_x, bw[0], bb[0], buf0,    BATCH, 512, 4);
    gemm(buf0,    bw[1], bb[1], buf1,    BATCH, 256, 512);
    gemm(buf1,    bw[2], bb[2], buf0,    BATCH, 128, 256);
    gemm(buf0,    bw[3], bb[3], bot_out, BATCH, 30,  128);

    // embedding gather + pairwise interaction -> R [16384, 381]
    gather_interact<<<dim3(BATCH / 8), blk, 0, stream>>>(bot_out, emb, lS_i, Rbuf);

    // top MLP: 381 -> 1024 -> 512 -> 512 -> 256 (ReLU), then 256 -> 1 (sigmoid)
    gemm(Rbuf, tw[0], tb[0], buf0, BATCH, 1024, R_DIM);
    gemm(buf0, tw[1], tb[1], buf1, BATCH, 512,  1024);
    gemm(buf1, tw[2], tb[2], buf0, BATCH, 512,  512);
    gemm(buf0, tw[3], tb[3], buf1, BATCH, 256,  512);
    matvec_sigmoid<<<dim3((BATCH + 255) / 256), blk, 0, stream>>>(
        buf1, tw[4], tb[4], out, BATCH, 256);
}
